// GRU_Decoder_87995289960798
// MI455X (gfx1250) — compile-verified, matrix operand order checked by reading, not run on hardware
//
#include <hip/hip_runtime.h>
#include <cstdint>

// ---------------------------------------------------------------------------
// GRU decoder B=128, T=512, H=1024, L=2 on MI455X (gfx1250, wave32, WMMA).
//   - weights + input converted to bf16 once (L2-resident: 25MB + 134MB)
//   - gi0 = input @ W_ih[0]^T + b precomputed as one big parallel WMMA GEMM
//   - ONE persistent kernel runs the whole 512-step recurrence with a
//     device-scope sense-reversing grid barrier (no 1024 kernel launches)
//   - hidden state carried fp32 (exact z*h) + bf16 (WMMA A operand): the
//     recurrent k-loop is pure global_load_b128 + v_wmma_f32_16x16x32_bf16
//   - block mapping: tn = blockIdx -> all 4 waves of a block read IDENTICAL
//     weight B-tiles, deduped in the WGP$ (~2x less L2 traffic per step)
// ---------------------------------------------------------------------------

typedef __attribute__((ext_vector_type(16))) __bf16 bf16x16;
typedef __attribute__((ext_vector_type(8)))  float  v8f;
typedef __attribute__((ext_vector_type(8)))  unsigned int u32x8;

__device__ __forceinline__ unsigned short f2bf(float f) {
    union { float f; uint32_t u; } v; v.f = f;
    uint32_t u = v.u;
    uint32_t r = u + 0x7FFFu + ((u >> 16) & 1u);   // RNE
    return (unsigned short)(r >> 16);
}
__device__ __forceinline__ uint32_t pk_bf16(float lo, float hi) {
    return (uint32_t)f2bf(lo) | ((uint32_t)f2bf(hi) << 16);
}

// 16x32 bf16 A/B tile from row-major bf16 matrix.
__device__ __forceinline__ bf16x16 ld_tile_bf16(const unsigned short* __restrict__ p,
                                                int ld, int rBase, int kk, int lane) {
    int r  = rBase + (lane & 15);
    int k0 = kk + ((lane & 16) ? 8 : 0);
    const unsigned short* q = p + (size_t)r * ld + k0;
    uint4 c0 = *(const uint4*)(q);
    uint4 c1 = *(const uint4*)(q + 16);
    u32x8 t;
    t[0] = c0.x; t[1] = c0.y; t[2] = c0.z; t[3] = c0.w;
    t[4] = c1.x; t[5] = c1.y; t[6] = c1.z; t[7] = c1.w;
    return __builtin_bit_cast(bf16x16, t);
}
// Fallback: tile from fp32 source, converted in-register.
__device__ __forceinline__ bf16x16 ld_tile_f32(const float* __restrict__ p,
                                               int ld, int rBase, int kk, int lane) {
    int r  = rBase + (lane & 15);
    int k0 = kk + ((lane & 16) ? 8 : 0);
    const float* q = p + (size_t)r * ld + k0;
    float4 a = *(const float4*)(q);
    float4 b = *(const float4*)(q + 4);
    float4 c = *(const float4*)(q + 16);
    float4 d = *(const float4*)(q + 20);
    u32x8 t;
    t[0] = pk_bf16(a.x, a.y); t[1] = pk_bf16(a.z, a.w);
    t[2] = pk_bf16(b.x, b.y); t[3] = pk_bf16(b.z, b.w);
    t[4] = pk_bf16(c.x, c.y); t[5] = pk_bf16(c.z, c.w);
    t[6] = pk_bf16(d.x, d.y); t[7] = pk_bf16(d.z, d.w);
    return __builtin_bit_cast(bf16x16, t);
}
__device__ __forceinline__ v8f wmma_bf16(bf16x16 a, bf16x16 b, v8f c) {
    return __builtin_amdgcn_wmma_f32_16x16x32_bf16(false, a, false, b,
                                                   (short)0, c, false, false);
}
__device__ __forceinline__ float sigmoidf_(float x) {
    return 1.0f / (1.0f + __expf(-x));
}

// ---------------------------------------------------------------------------
// small elementwise kernels
// ---------------------------------------------------------------------------
__global__ void cvt_bf16_kernel(const float* __restrict__ src,
                                unsigned short* __restrict__ dst, int n) {
    int i = blockIdx.x * blockDim.x + threadIdx.x;
    if (i < n) dst[i] = f2bf(src[i]);
}

__global__ void init_h_kernel(const float* __restrict__ enc,
                              float* __restrict__ h0f, float* __restrict__ h1f,
                              unsigned short* __restrict__ h0b,
                              unsigned short* __restrict__ h1b,
                              unsigned* __restrict__ barCnt,
                              unsigned* __restrict__ barGen, int n) {
    int i = blockIdx.x * blockDim.x + threadIdx.x;
    if (i == 0) { *barCnt = 0u; *barGen = 0u; }
    if (i < n) {
        float v = enc[i];
        h0f[i] = v; h1f[i] = v;
        unsigned short b = f2bf(v);
        h0b[i] = b; h1b[i] = b;
    }
}

// ---------------------------------------------------------------------------
// Time-parallel GEMM: G[M,N] = X[M,K] @ W[N,K]^T + bias[N]; M-blocked by 4.
// ---------------------------------------------------------------------------
template<bool USEXB>
__global__ void gi_gemm_kernel(const unsigned short* __restrict__ XB,
                               const float* __restrict__ XF,
                               const unsigned short* __restrict__ W,
                               const float* __restrict__ bias,
                               float* __restrict__ G,
                               int M, int N, int K) {
    int lane  = threadIdx.x & 31;
    int wave  = blockIdx.x * (blockDim.x >> 5) + (threadIdx.x >> 5);
    int nWv   = gridDim.x * (blockDim.x >> 5);
    int tMB   = M >> 6;
    int total = tMB * (N >> 4);
    for (int task = wave; task < total; task += nWv) {
        int tmb = task % tMB;           // m fastest: concurrent waves share B
        int tn  = task / tMB;
        int mB  = tmb << 6, nBase = tn << 4;
        v8f a0 = {}, a1 = {}, a2 = {}, a3 = {};
        for (int kk = 0; kk < K; kk += 32) {
            bf16x16 b = ld_tile_bf16(W, K, nBase, kk, lane);
            bf16x16 x0, x1, x2, x3;
            if constexpr (USEXB) {
                x0 = ld_tile_bf16(XB, K, mB,      kk, lane);
                x1 = ld_tile_bf16(XB, K, mB + 16, kk, lane);
                x2 = ld_tile_bf16(XB, K, mB + 32, kk, lane);
                x3 = ld_tile_bf16(XB, K, mB + 48, kk, lane);
            } else {
                x0 = ld_tile_f32(XF, K, mB,      kk, lane);
                x1 = ld_tile_f32(XF, K, mB + 16, kk, lane);
                x2 = ld_tile_f32(XF, K, mB + 32, kk, lane);
                x3 = ld_tile_f32(XF, K, mB + 48, kk, lane);
            }
            a0 = wmma_bf16(x0, b, a0);
            a1 = wmma_bf16(x1, b, a1);
            a2 = wmma_bf16(x2, b, a2);
            a3 = wmma_bf16(x3, b, a3);
        }
        int   n  = nBase + (lane & 15);
        float bv = bias[n];
        int   mo = (lane & 16) ? 8 : 0;
#pragma unroll
        for (int v = 0; v < 8; v++) {
            G[(size_t)(mB + mo + v) * N + n]      = a0[v] + bv;
            G[(size_t)(mB + 16 + mo + v) * N + n] = a1[v] + bv;
            G[(size_t)(mB + 32 + mo + v) * N + n] = a2[v] + bv;
            G[(size_t)(mB + 48 + mo + v) * N + n] = a3[v] + bv;
        }
    }
}

// ---------------------------------------------------------------------------
// device-scope sense-reversing grid barrier (all 64 blocks resident)
// ---------------------------------------------------------------------------
__device__ __forceinline__ void grid_barrier(unsigned* cnt, unsigned* gen, int nB) {
    __threadfence();               // release this block's stores device-wide
    __syncthreads();
    if (threadIdx.x == 0) {
        unsigned g = __hip_atomic_load(gen, __ATOMIC_RELAXED, __HIP_MEMORY_SCOPE_AGENT);
        unsigned a = __hip_atomic_fetch_add(cnt, 1u, __ATOMIC_ACQ_REL, __HIP_MEMORY_SCOPE_AGENT);
        if (a == (unsigned)nB - 1u) {
            __hip_atomic_store(cnt, 0u, __ATOMIC_RELAXED, __HIP_MEMORY_SCOPE_AGENT);
            __hip_atomic_fetch_add(gen, 1u, __ATOMIC_RELEASE, __HIP_MEMORY_SCOPE_AGENT);
        } else {
            while (__hip_atomic_load(gen, __ATOMIC_ACQUIRE, __HIP_MEMORY_SCOPE_AGENT) == g)
                __builtin_amdgcn_s_sleep(2);
        }
    }
    __syncthreads();
    __threadfence();               // acquire: don't read stale lines after barrier
}

// ---------------------------------------------------------------------------
// One fused GRU layer step: wave computes 2 m-tiles x 1 n-tile.
// MODE: 0 = gi precomputed, 1 = x from bf16 buffer, 2 = x from fp32 (convert)
// tn is per-BLOCK (blockIdx), tp per-wave: waves in a block share B-tiles.
// ---------------------------------------------------------------------------
template<int MODE, bool WOUT>
__device__ __forceinline__ void layer_step(
    int tp, int tn, int lane,
    const float* __restrict__ giPre, long long giLd,
    const unsigned short* __restrict__ xB, int xLd,
    const float* __restrict__ xF, int xLdF,
    const unsigned short* __restrict__ Wi,
    const unsigned short* __restrict__ Wh,
    const float* __restrict__ bi, const float* __restrict__ bh,
    const unsigned short* __restrict__ hB, const float* __restrict__ hF,
    float* __restrict__ hNewF, unsigned short* __restrict__ hNewB,
    float* __restrict__ out2, long long out2Ld) {
    const int Hc = 1024;
    int m0 = tp << 5, m1 = m0 + 16;
    int nBase = tn << 4;
    constexpr bool HASX = (MODE != 0);

    v8f iR0 = {}, iZ0 = {}, iN0 = {}, hR0 = {}, hZ0 = {}, hN0 = {};
    v8f iR1 = {}, iZ1 = {}, iN1 = {}, hR1 = {}, hZ1 = {}, hN1 = {};
    const size_t gstr = (size_t)Hc * Hc;

    for (int kk = 0; kk < Hc; kk += 32) {
        bf16x16 ah0 = ld_tile_bf16(hB, Hc, m0, kk, lane);
        bf16x16 ah1 = ld_tile_bf16(hB, Hc, m1, kk, lane);
        bf16x16 bR  = ld_tile_bf16(Wh,            Hc, nBase, kk, lane);
        bf16x16 bZ  = ld_tile_bf16(Wh + gstr,     Hc, nBase, kk, lane);
        bf16x16 bN  = ld_tile_bf16(Wh + 2 * gstr, Hc, nBase, kk, lane);
        hR0 = wmma_bf16(ah0, bR, hR0);  hR1 = wmma_bf16(ah1, bR, hR1);
        hZ0 = wmma_bf16(ah0, bZ, hZ0);  hZ1 = wmma_bf16(ah1, bZ, hZ1);
        hN0 = wmma_bf16(ah0, bN, hN0);  hN1 = wmma_bf16(ah1, bN, hN1);
        if constexpr (HASX) {
            bf16x16 ax0, ax1;
            if constexpr (MODE == 1) {
                ax0 = ld_tile_bf16(xB, xLd, m0, kk, lane);
                ax1 = ld_tile_bf16(xB, xLd, m1, kk, lane);
            } else {
                ax0 = ld_tile_f32(xF, xLdF, m0, kk, lane);
                ax1 = ld_tile_f32(xF, xLdF, m1, kk, lane);
            }
            bf16x16 cR = ld_tile_bf16(Wi,            Hc, nBase, kk, lane);
            bf16x16 cZ = ld_tile_bf16(Wi + gstr,     Hc, nBase, kk, lane);
            bf16x16 cN = ld_tile_bf16(Wi + 2 * gstr, Hc, nBase, kk, lane);
            iR0 = wmma_bf16(ax0, cR, iR0);  iR1 = wmma_bf16(ax1, cR, iR1);
            iZ0 = wmma_bf16(ax0, cZ, iZ0);  iZ1 = wmma_bf16(ax1, cZ, iZ1);
            iN0 = wmma_bf16(ax0, cN, iN0);  iN1 = wmma_bf16(ax1, cN, iN1);
        }
    }

    int   n   = nBase + (lane & 15);
    float bhr = bh[n], bhz = bh[Hc + n], bhn = bh[2 * Hc + n];
    float bir = 0.f, biz = 0.f, bin = 0.f;
    if constexpr (HASX) { bir = bi[n]; biz = bi[Hc + n]; bin = bi[2 * Hc + n]; }

#pragma unroll
    for (int p = 0; p < 2; p++) {
        int mBase = p ? m1 : m0;
        v8f aIR = p ? iR1 : iR0, aIZ = p ? iZ1 : iZ0, aIN = p ? iN1 : iN0;
        v8f aHR = p ? hR1 : hR0, aHZ = p ? hZ1 : hZ0, aHN = p ? hN1 : hN0;
        int mOff = mBase + ((lane & 16) ? 8 : 0);
#pragma unroll
        for (int v = 0; v < 8; v++) {
            int m = mOff + v;
            float ir, iz, inn;
            if constexpr (MODE == 0) {
                const float* g = giPre + (size_t)m * giLd;
                ir = g[n]; iz = g[Hc + n]; inn = g[2 * Hc + n];
            } else {
                ir = aIR[v] + bir; iz = aIZ[v] + biz; inn = aIN[v] + bin;
            }
            float hr = aHR[v] + bhr, hz = aHZ[v] + bhz, hn = aHN[v] + bhn;
            float ho = hF[(size_t)m * Hc + n];
            float r  = sigmoidf_(ir + hr);
            float z  = sigmoidf_(iz + hz);
            float nn = tanhf(inn + r * hn);
            float h  = (1.0f - z) * nn + z * ho;
            hNewF[(size_t)m * Hc + n] = h;
            hNewB[(size_t)m * Hc + n] = f2bf(h);
            if constexpr (WOUT) out2[(size_t)m * out2Ld + n] = h;
        }
    }
}

// ---------------------------------------------------------------------------
// Persistent recurrence kernel: whole T-loop in one launch.
// Grid: 64 blocks x 128 thr = 256 waves. tn = blockIdx (B-tile sharing in
// WGP$), tp = wave-in-block.
// ---------------------------------------------------------------------------
struct GruParams {
    const float* gi0;                  // [B*T, 3H] (row = b*T + t), MODE 0
    const unsigned short* xBf;         // bf16 [B*T, H], MODE 1
    const float* xF;                   // fp32 [B*T, H], MODE 2
    const unsigned short *wi0, *wh0, *wi1, *wh1;
    const float *bi0, *bh0, *bi1, *bh1;
    float *h0f0, *h0f1, *h1f0, *h1f1;
    unsigned short *h0b0, *h0b1, *h1b0, *h1b1;
    float* out;                        // [B, T, H]
    float* hFinal;                     // [B, H]
    unsigned *barCnt, *barGen;
    int nBlocks;
};

template<int MODE>
__global__ void gru_persistent_kernel(GruParams P) {
    const int T = 512, H = 1024, H3 = 3 * H, B = 128;
    int lane = threadIdx.x & 31;
    int tp   = threadIdx.x >> 5;       // 0..3   (m-tile pair)
    int tn   = blockIdx.x;             // 0..63  (n-tile; shared by the block)
    float*          h0f[2] = {P.h0f0, P.h0f1};
    float*          h1f[2] = {P.h1f0, P.h1f1};
    unsigned short* h0b[2] = {P.h0b0, P.h0b1};
    unsigned short* h1b[2] = {P.h1b0, P.h1b1};

    for (int t = 0; t < T; t++) {
        int cur = t & 1, nxt = cur ^ 1;
        // ---- layer 0 ----
        layer_step<MODE, false>(tp, tn, lane,
            P.gi0 + (size_t)t * H3, (long long)T * H3,
            P.xBf + (size_t)t * H, T * H,
            P.xF + (size_t)t * H, T * H,
            P.wi0, P.wh0, P.bi0, P.bh0,
            h0b[cur], h0f[cur], h0f[nxt], h0b[nxt], nullptr, 0);
        grid_barrier(P.barCnt, P.barGen, P.nBlocks);
        // ---- layer 1 (x = layer0 output, always bf16 buffer) ----
        layer_step<1, true>(tp, tn, lane,
            nullptr, 0,
            h0b[nxt], H,
            nullptr, 0,
            P.wi1, P.wh1, P.bi1, P.bh1,
            h1b[cur], h1f[cur], h1f[nxt], h1b[nxt],
            P.out + (size_t)t * H, (long long)T * H);
        grid_barrier(P.barCnt, P.barGen, P.nBlocks);
    }
    // final hidden state of last layer -> tail of d_out (after last barrier)
    const float* fin = (T & 1) ? h1f[1] : h1f[0];
    int tid = blockIdx.x * blockDim.x + threadIdx.x;
    int nT  = gridDim.x * blockDim.x;
    for (int i = tid; i < B * H; i += nT) P.hFinal[i] = fin[i];
}

// ---------------------------------------------------------------------------
extern "C" void kernel_launch(void* const* d_in, const int* in_sizes, int n_in,
                              void* d_out, int out_size, void* d_ws, size_t ws_size,
                              hipStream_t stream) {
    const int B = 128, T = 512, H = 1024;
    const int H3 = 3 * H;

    const float* input = (const float*)d_in[0];   // [B, T, H]
    const float* enc   = (const float*)d_in[1];   // [B, H]
    const float* Wih   = (const float*)d_in[2];   // [L, 3H, H]
    const float* Whh   = (const float*)d_in[3];
    const float* bih   = (const float*)d_in[4];   // [L, 3H]
    const float* bhh   = (const float*)d_in[5];
    float* out = (float*)d_out;                   // [B,T,H] then [B,H]

    const size_t nW = (size_t)2 * H3 * H;
    const size_t nX = (size_t)B * T * H;

    char* ws  = (char*)d_ws;
    size_t off = 0;
    auto alloc = [&](size_t bytes) -> char* {
        char* p = ws + off;
        off = (off + bytes + 255) & ~(size_t)255;
        return p;
    };

    unsigned short* wiB = (unsigned short*)alloc(nW * 2);
    unsigned short* whB = (unsigned short*)alloc(nW * 2);
    unsigned* barCnt = (unsigned*)alloc(8);
    unsigned* barGen = barCnt + 1;
    float*          h0f[2]; float* h1f[2];
    unsigned short* h0b[2]; unsigned short* h1b[2];
    for (int i = 0; i < 2; i++) {
        h0f[i] = (float*)alloc((size_t)B * H * 4);
        h1f[i] = (float*)alloc((size_t)B * H * 4);
        h0b[i] = (unsigned short*)alloc((size_t)B * H * 2);
        h1b[i] = (unsigned short*)alloc((size_t)B * H * 2);
    }
    bool useXB = (ws_size >= off + nX * 2);
    unsigned short* xBf = useXB ? (unsigned short*)alloc(nX * 2) : nullptr;
    const size_t giBytes = (size_t)B * T * H3 * 4;
    bool  pre  = (ws_size >= off + giBytes);
    float* gi0 = pre ? (float*)alloc(giBytes) : nullptr;

    // 1) conversions (parallel, bandwidth-bound)
    cvt_bf16_kernel<<<((int)nW + 255) / 256, 256, 0, stream>>>(Wih, wiB, (int)nW);
    cvt_bf16_kernel<<<((int)nW + 255) / 256, 256, 0, stream>>>(Whh, whB, (int)nW);
    if (useXB)
        cvt_bf16_kernel<<<((int)nX + 255) / 256, 256, 0, stream>>>(input, xBf, (int)nX);
    // 2) states + barrier init
    init_h_kernel<<<(B * H + 255) / 256, 256, 0, stream>>>(
        enc, h0f[0], h1f[0], h0b[0], h1b[0], barCnt, barGen, B * H);
    // 3) time-parallel precompute gi0
    if (pre) {
        if (useXB)
            gi_gemm_kernel<true ><<<4096, 256, 0, stream>>>(xBf, input, wiB, bih, gi0, B * T, H3, H);
        else
            gi_gemm_kernel<false><<<4096, 256, 0, stream>>>(xBf, input, wiB, bih, gi0, B * T, H3, H);
    }

    // 4) persistent recurrence (single launch, grid barriers between phases)
    const size_t lw = (size_t)H3 * H;
    GruParams P;
    P.gi0 = gi0; P.xBf = xBf; P.xF = input;
    P.wi0 = wiB; P.wh0 = whB; P.wi1 = wiB + lw; P.wh1 = whB + lw;
    P.bi0 = bih; P.bh0 = bhh; P.bi1 = bih + H3; P.bh1 = bhh + H3;
    P.h0f0 = h0f[0]; P.h0f1 = h0f[1]; P.h1f0 = h1f[0]; P.h1f1 = h1f[1];
    P.h0b0 = h0b[0]; P.h0b1 = h0b[1]; P.h1b0 = h1b[0]; P.h1b1 = h1b[1];
    P.out = out; P.hFinal = out + (size_t)B * T * H;
    P.barCnt = barCnt; P.barGen = barGen;
    P.nBlocks = 64;

    dim3 grid(64), block(128);
    if (pre)
        gru_persistent_kernel<0><<<grid, block, 0, stream>>>(P);
    else if (useXB)
        gru_persistent_kernel<1><<<grid, block, 0, stream>>>(P);
    else
        gru_persistent_kernel<2><<<grid, block, 0, stream>>>(P);
}
